// ClassificationBert_LSTM_5308579577982
// MI455X (gfx1250) — matrix operand order repaired and checked by reading, not compiled, not run on hardware
//
#include <hip/hip_runtime.h>
#include <hip/hip_bf16.h>

// ---------------- constants ----------------
constexpr int Bc = 32;       // batch
constexpr int Sc = 512;      // seq
constexpr int Ec = 768;      // embed
constexpr int Hc = 1024;     // hidden
constexpr int G4 = 4 * Hc;   // 4096 gates
constexpr int NLc = 5;       // labels
constexpr int Mc = Bc * Sc;  // 16384 rows (time-major: row = s*B + b)
constexpr int NWG_LSTM = Hc / 64;  // 16 persistent workgroups per LSTM layer

typedef __attribute__((ext_vector_type(16))) __bf16 v16bf;
typedef __attribute__((ext_vector_type(8)))  float  v8f;
typedef __attribute__((ext_vector_type(4)))  unsigned v4u;
typedef __attribute__((ext_vector_type(8)))  int      v8i;
typedef __attribute__((ext_vector_type(4)))  int      v4i;

union FragB16 { uint4 u[2]; v16bf v; };

__device__ inline v8f wmma_bf16(v16bf a, v16bf b, v8f c) {
  return __builtin_amdgcn_wmma_f32_16x16x32_bf16(
      /*neg_a=*/false, a, /*neg_b=*/false, b,
      /*c_mod=*/(short)0, c, /*reuse_a=*/false, /*reuse_b=*/false);
}

__device__ inline float sigf(float x) {
  x = fminf(fmaxf(x, -30.f), 30.f);
  return 1.f / (1.f + __expf(-x));
}
__device__ inline float tanhfast(float x) {
  x = fminf(fmaxf(x, -15.f), 15.f);
  float e = __expf(2.f * x);
  return (e - 1.f) / (e + 1.f);
}

__device__ inline unsigned lds_addr_u32(const void* p) {
  // generic LDS address: low 32 bits are the wave-relative LDS byte offset
  return (unsigned)(unsigned long long)p;
}

// TDM: DMA a [rows x cols] bf16 tile (contiguous rows, stride = cols) into LDS.
__device__ inline void tdm_load_bf16_2d(const void* gptr, unsigned lds_off,
                                        unsigned rows, unsigned cols) {
  unsigned long long ga = (unsigned long long)(size_t)gptr;
  v4u g0;
  g0[0] = 1u;                                        // count=1 (user mode)
  g0[1] = lds_off;                                   // lds_addr
  g0[2] = (unsigned)(ga & 0xffffffffu);              // global_addr[31:0]
  g0[3] = (unsigned)((ga >> 32) & 0x01ffffffu)       // global_addr[56:32]
          | (2u << 30);                              // type = 2 ("image")
  v8i g1;
  g1[0] = (int)(1u << 16);                           // data_size=1 (2B); wg_mask=0
  g1[1] = (int)((cols & 0xffffu) << 16);             // tensor_dim0 lo16
  g1[2] = (int)((cols >> 16) & 0xffffu) | (int)((rows & 0xffffu) << 16);
  g1[3] = (int)((rows >> 16) & 0xffffu) | (int)(cols << 16);  // tile_dim0 = cols
  g1[4] = (int)rows;                                 // tile_dim1 = rows; tile_dim2=0
  g1[5] = (int)cols;                                 // tensor_dim0_stride lo32
  g1[6] = 0;
  g1[7] = 0;
  v4i z4 = {0, 0, 0, 0};
  v8i z8 = {0, 0, 0, 0, 0, 0, 0, 0};
  __builtin_amdgcn_tensor_load_to_lds(g0, g1, z4, z4, z8, 0);
}

// ---------------- tiny utility kernels ----------------
__global__ void k_f32_to_bf16(const float* __restrict__ in,
                              __bf16* __restrict__ out, int n) {
  int i = blockIdx.x * blockDim.x + threadIdx.x;
  if (i < n) out[i] = (__bf16)in[i];
}

__global__ void k_add2(const float* __restrict__ a, const float* __restrict__ b,
                       float* __restrict__ o, int n) {
  int i = blockIdx.x * blockDim.x + threadIdx.x;
  if (i < n) o[i] = a[i] + b[i];
}

__global__ void k_zero_u32(unsigned* __restrict__ p, int n) {
  int i = blockIdx.x * blockDim.x + threadIdx.x;
  if (i < n) p[i] = 0u;
}

// ---------------- blend: hdn[s*B+b, e] (bf16, time-major) ----------------
__global__ void k_blend(const int* __restrict__ x,
                        const unsigned char* __restrict__ emo,
                        const float* __restrict__ ah,       // [B,S,E]
                        const float* __restrict__ wev,      // [V,E]
                        __bf16* __restrict__ out) {          // [S,B,E]
  int bs = blockIdx.x;            // b*S + s (input order)
  int b = bs / Sc, s = bs % Sc;
  int tok = x[bs];
  bool m = (s >= 1) && (emo[tok] != 0);
  const float* ap = ah + (size_t)bs * Ec;
  const float* wp = wev + (size_t)tok * Ec;
  __bf16* op = out + ((size_t)s * Bc + b) * Ec;
  for (int e = threadIdx.x; e < Ec; e += blockDim.x) {
    float v = ap[e];
    if (m) v = 0.5f * (v + wp[e]);
    op[e] = (__bf16)v;
  }
}

// ---- big bf16 WMMA GEMM, double-buffered async-to-LDS staging ------------
// C[M,N] = A[M,K] * W[N,K]^T + bias ; 128x128 block, 8 waves of 32x64.
__global__ __launch_bounds__(256) void k_gemm_bf16(
    const __bf16* __restrict__ A, const __bf16* __restrict__ W,
    const float* __restrict__ bias, float* __restrict__ C,
    int Ndim, int K) {
  __shared__ __bf16 sA[2][128 * 32];
  __shared__ __bf16 sW[2][128 * 32];
  const int tid = threadIdx.x;
  const int wave = tid / 32, lane = tid % 32;
  const int laneM = lane % 16, laneH = lane / 16;
  const int bm = blockIdx.x, bn = blockIdx.y;
  const int wm = wave % 4, wn = wave / 4;   // 4x2 wave grid -> 32x64 per wave

  v8f acc[2][4];
#pragma unroll
  for (int mt = 0; mt < 2; ++mt)
#pragma unroll
    for (int nt = 0; nt < 4; ++nt)
      acc[mt][nt] = (v8f){0.f, 0.f, 0.f, 0.f, 0.f, 0.f, 0.f, 0.f};

  const int ldr = tid / 2;        // 0..127
  const int ldp = (tid % 2) * 16; // 0 or 16 halves
  const unsigned ldsA0 = lds_addr_u32(&sA[0][ldr * 32 + ldp]);
  const unsigned ldsA1 = lds_addr_u32(&sA[1][ldr * 32 + ldp]);
  const unsigned ldsW0 = lds_addr_u32(&sW[0][ldr * 32 + ldp]);
  const unsigned ldsW1 = lds_addr_u32(&sW[1][ldr * 32 + ldp]);
  const unsigned long long baseA = (unsigned long long)(size_t)A;
  const unsigned long long baseW = (unsigned long long)(size_t)W;

  auto issue = [&](int kkTile, int buf) {
    unsigned la = buf ? ldsA1 : ldsA0;
    unsigned lw = buf ? ldsW1 : ldsW0;
    unsigned offA = (unsigned)(((size_t)(bm * 128 + ldr) * K + kkTile * 32 + ldp) * 2);
    unsigned offW = (unsigned)(((size_t)(bn * 128 + ldr) * K + kkTile * 32 + ldp) * 2);
    asm volatile(
        "global_load_async_to_lds_b128 %0, %1, %2\n\t"
        "global_load_async_to_lds_b128 %3, %4, %5"
        :
        : "v"(la), "v"(offA), "s"(baseA),
          "v"(lw), "v"(offW), "s"(baseW)
        : "memory");
  };

  const int nk = K / 32;
  issue(0, 0);
  asm volatile("s_wait_asynccnt 0" ::: "memory");
  __syncthreads();

  for (int kk = 0; kk < nk; ++kk) {
    const int cur = kk & 1;
    if (kk + 1 < nk) issue(kk + 1, cur ^ 1);   // overlap next tile with math

    const __bf16* cA = sA[cur];
    const __bf16* cW = sW[cur];
    v16bf af[2], bfg[4];
#pragma unroll
    for (int mt = 0; mt < 2; ++mt) {
      int row = wm * 32 + mt * 16 + laneM;
      FragB16 f;
      f.u[0] = *(const uint4*)(cA + row * 32 + 0  + laneH * 8);
      f.u[1] = *(const uint4*)(cA + row * 32 + 16 + laneH * 8);
      af[mt] = f.v;
    }
#pragma unroll
    for (int nt = 0; nt < 4; ++nt) {
      int col = wn * 64 + nt * 16 + laneM;
      FragB16 f;
      const uint4* p = (const uint4*)(cW + col * 32 + laneH * 8);
      f.u[0] = p[0]; f.u[1] = p[2];
      bfg[nt] = f.v;
    }
#pragma unroll
    for (int mt = 0; mt < 2; ++mt)
#pragma unroll
      for (int nt = 0; nt < 4; ++nt)
        acc[mt][nt] = wmma_bf16(af[mt], bfg[nt], acc[mt][nt]);

    if (kk + 1 < nk) asm volatile("s_wait_asynccnt 0" ::: "memory");
    __syncthreads();
  }

#pragma unroll
  for (int mt = 0; mt < 2; ++mt)
#pragma unroll
    for (int nt = 0; nt < 4; ++nt) {
      int col = bn * 128 + wn * 64 + nt * 16 + laneM;
      float bb = bias[col];
#pragma unroll
      for (int v = 0; v < 8; ++v) {
        int row = bm * 128 + wm * 32 + mt * 16 + v + 8 * laneH;
        C[(size_t)row * Ndim + col] = acc[mt][nt][v] + bb;
      }
    }
}

// ---- persistent full-layer LSTM: 16 WGs, internal t-loop, c in registers --
// WG bx owns gate columns [bx*64, bx*64+64) of all 4 gates; grid barrier per
// step via per-step atomic counters. h_{t-1} staged into LDS by TDM each step.
__global__ __launch_bounds__(256) void k_lstm_layer(
    const __bf16* __restrict__ Whh,   // [4H, H] bf16
    const float*  __restrict__ G,     // [S, B, 4H] input-proj preacts
    const __bf16* __restrict__ hz,    // [B, H] zeros (h_{-1})
    __bf16* __restrict__ outb,        // [S, B, H] bf16 h sequence
    float*  __restrict__ outf,        // [S, B, H] f32 or nullptr
    int* __restrict__ cnt) {          // [S] zeroed arrival counters
  extern __shared__ char smem[];
  __bf16* hs = (__bf16*)smem;         // 32*1024 bf16 = 64 KB
  const int tid = threadIdx.x;
  const int wave = tid / 32, lane = tid % 32;
  const int laneM = lane % 16, laneH = lane / 16;
  const int mt = wave / 4;                 // 0..1 : batch rows 0-15 / 16-31
  const int nq = wave % 4;                 // 0..3 : 16-col chunk
  const int jbase = blockIdx.x * 64 + nq * 16;
  const int j = jbase + laneM;
  const unsigned lds_off = lds_addr_u32(hs);

  float creg[8];                           // cell state lives in registers
#pragma unroll
  for (int v = 0; v < 8; ++v) creg[v] = 0.f;

  for (int t = 0; t < Sc; ++t) {
    const __bf16* hprev = t ? (outb + (size_t)(t - 1) * Bc * Hc) : hz;
    if (tid < 32) {
      tdm_load_bf16_2d(hprev, lds_off, Bc, Hc);
      __builtin_amdgcn_s_wait_tensorcnt(0);
    }
    __syncthreads();

    const float* Gt = G + (size_t)t * Bc * G4;
    v8f acc[4];
#pragma unroll
    for (int g = 0; g < 4; ++g) {
      int col = g * Hc + jbase + laneM;
      v8f a;
#pragma unroll
      for (int v = 0; v < 8; ++v) {
        int row = mt * 16 + v + 8 * laneH;
        a[v] = Gt[(size_t)row * G4 + col];
      }
      acc[g] = a;
    }

    for (int k0 = 0; k0 < Hc; k0 += 32) {
      int row = mt * 16 + laneM;
      FragB16 fa;
      fa.u[0] = *(const uint4*)(hs + row * Hc + k0 + 0  + laneH * 8);
      fa.u[1] = *(const uint4*)(hs + row * Hc + k0 + 16 + laneH * 8);
#pragma unroll
      for (int g = 0; g < 4; ++g) {
        int n = g * Hc + jbase + laneM;
        FragB16 fb;
        const uint4* p = (const uint4*)(Whh + (size_t)n * Hc + k0 + laneH * 8);
        fb.u[0] = p[0]; fb.u[1] = p[2];
        acc[g] = wmma_bf16(fa.v, fb.v, acc[g]);
      }
    }

    float hn[8];
#pragma unroll
    for (int v = 0; v < 8; ++v) {
      float ig = sigf(acc[0][v]);
      float fg = sigf(acc[1][v]);
      float gg = tanhfast(acc[2][v]);
      float og = sigf(acc[3][v]);
      creg[v] = fg * creg[v] + ig * gg;
      hn[v] = og * tanhfast(creg[v]);
    }
    __bf16* ho = outb + (size_t)t * Bc * Hc;
#pragma unroll
    for (int v = 0; v < 8; ++v) {
      int bidx = mt * 16 + v + 8 * laneH;
      ho[(size_t)bidx * Hc + j] = (__bf16)hn[v];
    }
    if (outf) {
      float* hf = outf + (size_t)t * Bc * Hc;
#pragma unroll
      for (int v = 0; v < 8; ++v) {
        int bidx = mt * 16 + v + 8 * laneH;
        hf[(size_t)bidx * Hc + j] = hn[v];
      }
    }

    // ---- grid barrier: all 16 WGs finish step t before anyone reads h_t
    __threadfence();                       // make this thread's h stores visible
    __syncthreads();
    if (tid == 0) {
      __hip_atomic_fetch_add(&cnt[t], 1, __ATOMIC_RELEASE,
                             __HIP_MEMORY_SCOPE_AGENT);
      while (__hip_atomic_load(&cnt[t], __ATOMIC_ACQUIRE,
                               __HIP_MEMORY_SCOPE_AGENT) < NWG_LSTM) {
        __builtin_amdgcn_s_sleep(1);
      }
    }
    __syncthreads();
  }
}

// ---------------- attention score: score[b,s] ----------------
__global__ __launch_bounds__(128) void k_attn_score(
    const float* __restrict__ out2f,  // [S,B,H]
    const float* __restrict__ W1, const float* __restrict__ b1,
    const float* __restrict__ W2, const float* __restrict__ b2,
    float* __restrict__ score) {      // [B,S]
  __shared__ float xrow[Hc];
  __shared__ float red[128];
  int bs = blockIdx.x;
  int b = bs / Sc, s = bs % Sc;
  const float* xp = out2f + ((size_t)s * Bc + b) * Hc;
  for (int jj = threadIdx.x; jj < Hc; jj += 128) xrow[jj] = xp[jj];
  __syncthreads();
  float val = 0.f;
  int r = threadIdx.x;
  if (r < 100) {
    float accv = b1[r];
    const float* wr = W1 + (size_t)r * Hc;
    for (int jj = 0; jj < Hc; ++jj) accv += xrow[jj] * wr[jj];
    val = tanhfast(accv) * W2[r];
  }
  red[threadIdx.x] = val;
  __syncthreads();
  for (int st = 64; st > 0; st >>= 1) {
    if (threadIdx.x < st) red[threadIdx.x] += red[threadIdx.x + st];
    __syncthreads();
  }
  if (threadIdx.x == 0) score[(size_t)b * Sc + s] = red[0] + b2[0];
}

// ---------------- softmax over s per b ----------------
__global__ __launch_bounds__(256) void k_softmax(const float* __restrict__ score,
                                                 float* __restrict__ wout) {
  __shared__ float red[256];
  int b = blockIdx.x;
  const float* sp = score + (size_t)b * Sc;
  float mx = -1e30f;
  for (int s = threadIdx.x; s < Sc; s += 256) mx = fmaxf(mx, sp[s]);
  red[threadIdx.x] = mx;
  __syncthreads();
  for (int st = 128; st > 0; st >>= 1) {
    if (threadIdx.x < st) red[threadIdx.x] = fmaxf(red[threadIdx.x], red[threadIdx.x + st]);
    __syncthreads();
  }
  mx = red[0];
  __syncthreads();
  float sum = 0.f;
  for (int s = threadIdx.x; s < Sc; s += 256) sum += __expf(sp[s] - mx);
  red[threadIdx.x] = sum;
  __syncthreads();
  for (int st = 128; st > 0; st >>= 1) {
    if (threadIdx.x < st) red[threadIdx.x] += red[threadIdx.x + st];
    __syncthreads();
  }
  float inv = 1.f / red[0];
  for (int s = threadIdx.x; s < Sc; s += 256)
    wout[(size_t)b * Sc + s] = __expf(sp[s] - mx) * inv;
}

// ---------------- weighted sum + final linear ----------------
__global__ __launch_bounds__(256) void k_final(
    const float* __restrict__ out2f,  // [S,B,H]
    const float* __restrict__ w,      // [B,S]
    const float* __restrict__ Wf, const float* __restrict__ bfv,
    float* __restrict__ logits) {     // [B,NL]
  __shared__ float sent[Hc];
  int b = blockIdx.x;
  const float* wb = w + (size_t)b * Sc;
#pragma unroll
  for (int q = 0; q < 4; ++q) {
    int h = threadIdx.x + q * 256;
    float accv = 0.f;
    for (int s = 0; s < Sc; ++s)
      accv += wb[s] * out2f[((size_t)s * Bc + b) * Hc + h];
    sent[h] = accv;
  }
  __syncthreads();
  if (threadIdx.x < NLc) {
    float accv = bfv[threadIdx.x];
    const float* wr = Wf + (size_t)threadIdx.x * Hc;
    for (int h = 0; h < Hc; ++h) accv += sent[h] * wr[h];
    logits[(size_t)b * NLc + threadIdx.x] = accv;
  }
}

// ---------------- host orchestration ----------------
extern "C" void kernel_launch(void* const* d_in, const int* in_sizes, int n_in,
                              void* d_out, int out_size, void* d_ws, size_t ws_size,
                              hipStream_t stream) {
  (void)in_sizes; (void)n_in; (void)out_size; (void)ws_size;
  const int*   x    = (const int*)d_in[0];
  const unsigned char* emo = (const unsigned char*)d_in[1];
  const float* ah   = (const float*)d_in[2];
  const float* wev  = (const float*)d_in[3];
  const float* Wih0 = (const float*)d_in[4];
  const float* Whh0 = (const float*)d_in[5];
  const float* bih0 = (const float*)d_in[6];
  const float* bhh0 = (const float*)d_in[7];
  const float* Wih1 = (const float*)d_in[8];
  const float* Whh1 = (const float*)d_in[9];
  const float* bih1 = (const float*)d_in[10];
  const float* bhh1 = (const float*)d_in[11];
  const float* W1   = (const float*)d_in[12];
  const float* b1   = (const float*)d_in[13];
  const float* W2   = (const float*)d_in[14];
  const float* b2   = (const float*)d_in[15];
  const float* Wf   = (const float*)d_in[16];
  const float* bfv  = (const float*)d_in[17];

  char* ws = (char*)d_ws;
  auto alloc = [&](size_t bytes) -> char* {
    char* p = ws;
    ws += (bytes + 255) & ~(size_t)255;
    return p;
  };
  __bf16* hdnb   = (__bf16*)alloc((size_t)Mc * Ec * 2);
  __bf16* wih0b  = (__bf16*)alloc((size_t)G4 * Ec * 2);
  __bf16* whh0b  = (__bf16*)alloc((size_t)G4 * Hc * 2);
  __bf16* wih1b  = (__bf16*)alloc((size_t)G4 * Hc * 2);
  __bf16* whh1b  = (__bf16*)alloc((size_t)G4 * Hc * 2);
  float*  biasg0 = (float*)alloc((size_t)G4 * 4);
  float*  biasg1 = (float*)alloc((size_t)G4 * 4);
  float*  G      = (float*)alloc((size_t)Mc * G4 * 4);      // gate preacts (reused per layer)
  __bf16* out1b  = (__bf16*)alloc((size_t)Mc * Hc * 2);     // [S,B,H] layer-0 h seq
  __bf16* out2b  = (__bf16*)alloc((size_t)Mc * Hc * 2);     // [S,B,H] layer-1 h seq (bf16 chain)
  float*  out2f  = (float*)alloc((size_t)Mc * Hc * 4);      // [S,B,H] f32 for attention
  __bf16* hz     = (__bf16*)alloc((size_t)Bc * Hc * 2);     // zero h0
  float*  score  = (float*)alloc((size_t)Bc * Sc * 4);
  int*    cnt    = (int*)alloc((size_t)Sc * 4);             // grid-barrier counters

  auto cdiv = [](int a, int b) { return (a + b - 1) / b; };

  // 1) weight conversion + fused bias
  k_f32_to_bf16<<<cdiv(G4 * Ec, 256), 256, 0, stream>>>(Wih0, wih0b, G4 * Ec);
  k_f32_to_bf16<<<cdiv(G4 * Hc, 256), 256, 0, stream>>>(Whh0, whh0b, G4 * Hc);
  k_f32_to_bf16<<<cdiv(G4 * Hc, 256), 256, 0, stream>>>(Wih1, wih1b, G4 * Hc);
  k_f32_to_bf16<<<cdiv(G4 * Hc, 256), 256, 0, stream>>>(Whh1, whh1b, G4 * Hc);
  k_add2<<<cdiv(G4, 256), 256, 0, stream>>>(bih0, bhh0, biasg0, G4);
  k_add2<<<cdiv(G4, 256), 256, 0, stream>>>(bih1, bhh1, biasg1, G4);
  k_zero_u32<<<cdiv(Bc * Hc / 2, 256), 256, 0, stream>>>((unsigned*)hz, Bc * Hc / 2);

  // 2) blend to time-major bf16
  k_blend<<<Mc, 256, 0, stream>>>(x, emo, ah, wev, hdnb);

  // 3) layer-0 input projection: G = hdn @ Wih0^T + (bih0+bhh0)
  dim3 gdim(Mc / 128, G4 / 128);
  k_gemm_bf16<<<gdim, 256, 0, stream>>>(hdnb, wih0b, biasg0, G, G4, Ec);

  // 4) layer-0 recurrence: one persistent kernel, 16 co-resident WGs
  k_zero_u32<<<cdiv(Sc, 256), 256, 0, stream>>>((unsigned*)cnt, Sc);
  k_lstm_layer<<<NWG_LSTM, 256, Bc * Hc * 2, stream>>>(
      whh0b, G, hz, out1b, nullptr, cnt);

  // 5) layer-1 input projection: G = out1 @ Wih1^T + (bih1+bhh1)
  k_gemm_bf16<<<gdim, 256, 0, stream>>>(out1b, wih1b, biasg1, G, G4, Hc);

  // 6) layer-1 recurrence (also writes f32 output sequence)
  k_zero_u32<<<cdiv(Sc, 256), 256, 0, stream>>>((unsigned*)cnt, Sc);
  k_lstm_layer<<<NWG_LSTM, 256, Bc * Hc * 2, stream>>>(
      whh1b, G, hz, out2b, out2f, cnt);

  // 7) attention -> softmax -> pooled logits
  float* logits = (float*)d_out;             // [B,NL]
  float* wout = logits + Bc * NLc;           // [B,S]
  k_attn_score<<<Mc, 128, 0, stream>>>(out2f, W1, b1, W2, b2, score);
  k_softmax<<<Bc, 256, 0, stream>>>(score, wout);
  k_final<<<Bc, 256, 0, stream>>>(out2f, wout, Wf, bfv, logits);
}
